// HGT_87514253623568
// MI455X (gfx1250) — matrix operand order repaired and checked by reading, not compiled, not run on hardware
//
#include <hip/hip_runtime.h>
#include <hip/hip_bf16.h>
#include <cmath>

typedef __attribute__((ext_vector_type(16))) __bf16    v16bf;
typedef __attribute__((ext_vector_type(8)))  float     v8f;
typedef __attribute__((ext_vector_type(4)))  uint32_t  u32x4;
typedef __attribute__((ext_vector_type(8)))  uint32_t  u32x8;

#define NA_N   20000
#define NP_N   40000
#define EWB_N  150000
#define EW_N   150000
#define EC_N   300000
#define DIM    256

// ---------------------------------------------------------------------------
// hi/lo bf16 split of a K-pair into fragment slots 2r, 2r+1.
// ---------------------------------------------------------------------------
static __device__ __forceinline__ void pack_pair(float f0, float f1,
                                                 v16bf& hi, v16bf& lo, int r) {
  __bf16 h0 = (__bf16)f0, h1 = (__bf16)f1;
  hi[2 * r]     = h0;
  hi[2 * r + 1] = h1;
  lo[2 * r]     = (__bf16)(f0 - (float)h0);
  lo[2 * r + 1] = (__bf16)(f1 - (float)h1);
}

// ---------------------------------------------------------------------------
// Pre-pack a weight panel W[256 x NCOLS] (f32, row-major) into bf16 hi/lo
// dword arrays in WMMA B-fragment order: index = ((kk*NTILES+nt)*32+lane)*8+r
// dword = {bf16(W[kb+1][n]) , bf16(W[kb][n])}, kb = kk*32 + 16*(r>=4) + 8*(lane>=16) + 2*(r&3)
// grid: 8*NTILES blocks, 256 threads (thread = (lane, r)).
// ---------------------------------------------------------------------------
__global__ __launch_bounds__(256)
void pack_weight_kernel(const float* __restrict__ W, uint32_t* __restrict__ hi,
                        uint32_t* __restrict__ lo, int NTILES) {
  const int nt   = blockIdx.x % NTILES;
  const int kk   = blockIdx.x / NTILES;
  const int lane = threadIdx.x >> 3;
  const int r    = threadIdx.x & 7;
  const int NC   = NTILES * 16;
  const int n    = nt * 16 + (lane & 15);
  const int kb   = kk * 32 + ((r & 4) ? 16 : 0) + ((lane & 16) ? 8 : 0) + 2 * (r & 3);
  const float f0 = W[(size_t)kb * NC + n];
  const float f1 = W[(size_t)(kb + 1) * NC + n];
  const __bf16 h0 = (__bf16)f0, h1 = (__bf16)f1;
  const __bf16 l0 = (__bf16)(f0 - (float)h0), l1 = (__bf16)(f1 - (float)h1);
  union { __bf16 b[2]; uint32_t u; } ph, pl;
  ph.b[0] = h0; ph.b[1] = h1;
  pl.b[0] = l0; pl.b[1] = l1;
  const size_t idx = ((size_t)blockIdx.x * 32 + lane) * 8 + r;
  hi[idx] = ph.u;
  lo[idx] = pl.u;
}

// ---------------------------------------------------------------------------
// WMMA GEMM: C[M x NCOLS] = epi(A[M x 256]*a_scale @ W + bias)
//  - A tile (32 x 256 f32) DMA'd into LDS by the Tensor Data Mover (one
//    tensor_load_to_lds per block, TENSORcnt-synchronized).
//  - B fragments loaded pre-packed (bf16 hi/lo) straight from global (L2).
//  - 3x v_wmma_f32_16x16x32_bf16 per tile-step (2-term bf16 split mul).
// Block: 256 threads (8 wave32). EPI: 0 = none, 1 = exact GELU.
// Requires M % 32 == 0 (true for all call sites).
// ---------------------------------------------------------------------------
template <int NCOLS, int EPI>
__global__ __launch_bounds__(256)
void wmma_gemm(const float* __restrict__ A,
               const uint32_t* __restrict__ Whi, const uint32_t* __restrict__ Wlo,
               const float* __restrict__ bias, float* __restrict__ C,
               int M, float a_scale) {
  constexpr int NTILES = NCOLS / 16;       // 16 (N=256) or 4 (N=64)
  constexpr int TPW    = (2 * NTILES) / 8; // C tiles per wave

  __shared__ float As[32][DIM];            // 32 KB, LDS offset 0

  const int tid  = threadIdx.x;
  const int wave = tid >> 5;
  const int lane = tid & 31;
  const int m0   = blockIdx.x * 32;
  const int mt   = (wave * TPW) / NTILES;  // constant per wave

  // --- TDM: async-load the whole A tile (32 rows x 256 cols, f32) to LDS ---
  if (wave == 0) {
    const uint64_t ga = (uint64_t)(uintptr_t)A + (uint64_t)m0 * (DIM * 4);
    u32x4 g0;
    g0[0] = 1u;                                   // count=1, user mode
    g0[1] = 0u;                                   // lds_addr = 0 (As at base)
    g0[2] = (uint32_t)ga;                         // global_addr[31:0]
    g0[3] = (uint32_t)(ga >> 32) | (2u << 30);    // global_addr[56:32] | type=2
    u32x8 g1;
    g1[0] = 2u << 16;                             // data_size = 4 bytes
    g1[1] = ((uint32_t)DIM & 0xFFFFu) << 16;      // tensor_dim0 = 256 (lo16)
    g1[2] = ((uint32_t)M & 0xFFFFu) << 16;        // dim0 hi16=0 | tensor_dim1 lo16
    g1[3] = ((uint32_t)M >> 16) | ((uint32_t)DIM << 16);  // dim1 hi16 | tile_dim0=256
    g1[4] = 32u;                                  // tile_dim1 = 32, tile_dim2 = 0
    g1[5] = (uint32_t)DIM;                        // tensor_dim0_stride = 256
    g1[6] = 0u;
    g1[7] = 0u;
    asm volatile("tensor_load_to_lds %0, %1" :: "s"(g0), "s"(g1) : "memory");
    __builtin_amdgcn_s_wait_tensorcnt(0);
  }
  __syncthreads();

  v8f acc[TPW];
  for (int i = 0; i < TPW; ++i)
    for (int r = 0; r < 8; ++r) acc[i][r] = 0.0f;

  const int arow = mt * 16 + (lane & 15);
  for (int kk = 0; kk < 8; ++kk) {
    const int k0 = kk * 32;

    // --- pack A fragment (16x32, ISA layout: half-lane K offset 8) ---
    v16bf ahi, alo;
    for (int r = 0; r < 8; ++r) {
      const int base = (r & 4) ? 16 : 0;
      const int ka   = k0 + base + ((lane & 16) ? 8 : 0) + 2 * (r & 3);
      pack_pair(As[arow][ka] * a_scale, As[arow][ka + 1] * a_scale, ahi, alo, r);
    }

    for (int i = 0; i < TPW; ++i) {
      const int nt = (wave * TPW + i) % NTILES;
      // --- B fragments: pre-packed, straight from global (L2-resident) ---
      const size_t fidx = ((size_t)(kk * NTILES + nt) * 32 + lane) * 8;
      const v16bf bhi = *(const v16bf*)(Whi + fidx);
      const v16bf blo = *(const v16bf*)(Wlo + fidx);
      acc[i] = __builtin_amdgcn_wmma_f32_16x16x32_bf16(
          false, ahi, false, bhi, (short)0, acc[i], false, false);
      acc[i] = __builtin_amdgcn_wmma_f32_16x16x32_bf16(
          false, alo, false, bhi, (short)0, acc[i], false, false);
      acc[i] = __builtin_amdgcn_wmma_f32_16x16x32_bf16(
          false, ahi, false, blo, (short)0, acc[i], false, false);
    }
  }

  // --- epilogue: bias (+ optional exact GELU), store ---
  for (int i = 0; i < TPW; ++i) {
    const int nt = (wave * TPW + i) % NTILES;
    const int gc = nt * 16 + (lane & 15);
    const float bv = bias[gc];
    for (int r = 0; r < 8; ++r) {
      const int gm = m0 + mt * 16 + r + ((lane & 16) ? 8 : 0);
      if (gm < M) {
        float v = acc[i][r] + bv;
        if (EPI == 1) v = 0.5f * v * (1.0f + erff(v * 0.70710678118654752f));
        C[(size_t)gm * NCOLS + gc] = v;
      }
    }
  }
}

// ---------------------------------------------------------------------------
// Combine per-head relation weights into full projection:
//   Wc[i][j] = sum_d projW[i][h*64+d] * wrel[h][d][j%64],  h = j/64
//   bc[j]    = sum_d projB[h*64+d]    * wrel[h][d][j%64]
// grid: 257 blocks (block 256 == bias), block: 256 threads (one per column).
// ---------------------------------------------------------------------------
__global__ __launch_bounds__(256)
void combine_kernel(const float* __restrict__ pW, const float* __restrict__ pB,
                    const float* __restrict__ wrel, float* __restrict__ Wc,
                    float* __restrict__ bc) {
  const int j  = threadIdx.x;
  const int i  = blockIdx.x;
  const int h  = j >> 6;
  const int jj = j & 63;
  const float* wcol = wrel + h * 4096 + jj;  // stride 64 over d
  if (i < 256) {
    const float* arow = pW + (size_t)i * DIM + h * 64;
    float s = 0.f;
    for (int d = 0; d < 64; ++d) s += arow[d] * wcol[d * 64];
    Wc[(size_t)i * DIM + j] = s;
  } else {
    const float* arow = pB + h * 64;
    float s = 0.f;
    for (int d = 0; d < 64; ++d) s += arow[d] * wcol[d * 64];
    bc[j] = s;
  }
}

// ---------------------------------------------------------------------------
// Edge kernel: one wave32 per edge; lane covers 8 floats (head = lane/8).
//   score_h = dot(q[dst,h,:], k[src,h,:]) * w_pri[h] / 8
//   attn_h  = sigmoid(score_h) * ew[e];  agg[dst] += m[src] * attn (atomics)
// ---------------------------------------------------------------------------
__global__ __launch_bounds__(256)
void edge_kernel(const float* __restrict__ q, const float* __restrict__ k,
                 const float* __restrict__ m, const int* __restrict__ src,
                 const int* __restrict__ dst, const float* __restrict__ ew,
                 const float* __restrict__ w_pri, float* __restrict__ agg,
                 int E) {
  const int wave = threadIdx.x >> 5;
  const int lane = threadIdx.x & 31;
  const int e = blockIdx.x * 8 + wave;
  if (e >= E) return;

  const int s = src[e];
  const int d = dst[e];
  const float* qp = q + (size_t)d * DIM + lane * 8;
  const float* kp = k + (size_t)s * DIM + lane * 8;

  const float4 q0  = *(const float4*)qp;
  const float4 q1  = *(const float4*)(qp + 4);
  const float4 kv0 = *(const float4*)kp;
  const float4 kv1 = *(const float4*)(kp + 4);

  float part = q0.x * kv0.x + q0.y * kv0.y + q0.z * kv0.z + q0.w * kv0.w +
               q1.x * kv1.x + q1.y * kv1.y + q1.z * kv1.z + q1.w * kv1.w;
  part += __shfl_xor(part, 1, 8);
  part += __shfl_xor(part, 2, 8);
  part += __shfl_xor(part, 4, 8);

  const int h = lane >> 3;
  const float score = part * w_pri[h] * 0.125f;   // / sqrt(DK=64)
  const float attn  = ew[e] / (1.0f + __expf(-score));

  const float* mp = m + (size_t)s * DIM + lane * 8;
  float* ap = agg + (size_t)d * DIM + lane * 8;
  const float4 m0 = *(const float4*)mp;
  const float4 m1 = *(const float4*)(mp + 4);
  unsafeAtomicAdd(ap + 0, m0.x * attn);
  unsafeAtomicAdd(ap + 1, m0.y * attn);
  unsafeAtomicAdd(ap + 2, m0.z * attn);
  unsafeAtomicAdd(ap + 3, m0.w * attn);
  unsafeAtomicAdd(ap + 4, m1.x * attn);
  unsafeAtomicAdd(ap + 5, m1.y * attn);
  unsafeAtomicAdd(ap + 6, m1.z * attn);
  unsafeAtomicAdd(ap + 7, m1.w * attn);
}

// ---------------------------------------------------------------------------
// Skip + LayerNorm: one wave32 per node row (8 floats per lane).
// ---------------------------------------------------------------------------
__global__ __launch_bounds__(256)
void skip_ln_kernel(const float* __restrict__ tr, const float* __restrict__ feat,
                    const float* __restrict__ skip_p, const float* __restrict__ g,
                    const float* __restrict__ b, float* __restrict__ out, int N) {
  const int wave = threadIdx.x >> 5;
  const int lane = threadIdx.x & 31;
  const int n = blockIdx.x * 8 + wave;
  if (n >= N) return;

  const float alpha = 1.0f / (1.0f + __expf(-skip_p[0]));
  const float* tp = tr + (size_t)n * DIM + lane * 8;
  const float* fp = feat + (size_t)n * DIM + lane * 8;

  float o[8];
  float s = 0.f;
  for (int j = 0; j < 8; ++j) {
    o[j] = tp[j] * alpha + fp[j] * (1.0f - alpha);
    s += o[j];
  }
  for (int msk = 1; msk < 32; msk <<= 1) s += __shfl_xor(s, msk, 32);
  const float mu = s * (1.0f / 256.0f);

  float v = 0.f;
  for (int j = 0; j < 8; ++j) {
    const float dd = o[j] - mu;
    v += dd * dd;
  }
  for (int msk = 1; msk < 32; msk <<= 1) v += __shfl_xor(v, msk, 32);
  const float rs = rsqrtf(v * (1.0f / 256.0f) + 1e-5f);

  float* op = out + (size_t)n * DIM + lane * 8;
  const int c0 = lane * 8;
  for (int j = 0; j < 8; ++j)
    op[j] = (o[j] - mu) * rs * g[c0 + j] + b[c0 + j];
}

// ---------------------------------------------------------------------------
// Host orchestration
// ---------------------------------------------------------------------------
extern "C" void kernel_launch(void* const* d_in, const int* in_sizes, int n_in,
                              void* d_out, int out_size, void* d_ws, size_t ws_size,
                              hipStream_t stream) {
  (void)in_sizes; (void)n_in; (void)out_size; (void)ws_size;

  const float* x_a   = (const float*)d_in[0];
  const float* x_p   = (const float*)d_in[1];
  const float* ew_wb = (const float*)d_in[2];
  const float* ew_w  = (const float*)d_in[3];
  const float* ew_c  = (const float*)d_in[4];
  const float* adW   = (const float*)d_in[5];
  const float* adB   = (const float*)d_in[6];
  const float* kW    = (const float*)d_in[7];
  const float* kb    = (const float*)d_in[8];
  const float* qW    = (const float*)d_in[9];
  const float* qb    = (const float*)d_in[10];
  const float* mW    = (const float*)d_in[11];
  const float* mb    = (const float*)d_in[12];
  const float* aW    = (const float*)d_in[13];
  const float* ab    = (const float*)d_in[14];
  const float* wpri  = (const float*)d_in[15];
  const float* watt  = (const float*)d_in[16];
  const float* wmsg  = (const float*)d_in[17];
  const float* skp   = (const float*)d_in[18];
  const float* lng   = (const float*)d_in[19];
  const float* lnb   = (const float*)d_in[20];
  const float* outW  = (const float*)d_in[21];
  const float* outB  = (const float*)d_in[22];
  const int* src_wb  = (const int*)d_in[23];
  const int* dst_wb  = (const int*)d_in[24];
  const int* src_w   = (const int*)d_in[25];
  const int* dst_w   = (const int*)d_in[26];
  const int* src_c   = (const int*)d_in[27];
  const int* dst_c   = (const int*)d_in[28];

  const size_t NAF = (size_t)NA_N * DIM;   // 5,120,000
  const size_t NPF = (size_t)NP_N * DIM;   // 10,240,000

  float* ws = (float*)d_ws;
  size_t off = 0;
  float* f0A  = ws + off; off += NAF;
  float* f0P  = ws + off; off += NPF;
  float* f1A  = ws + off; off += NAF;
  float* f1P  = ws + off; off += NPF;
  float* qA   = ws + off; off += NAF;
  float* qP   = ws + off; off += NPF;
  float* kbuf = ws + off; off += NPF;      // sized for larger (paper) type
  float* mbuf = ws + off; off += NPF;
  float* aggA = ws + off; off += NAF;
  float* aggP = ws + off; off += NPF;
  float* cw   = ws + off; off += 12 * 65536;  // combined K/M weights (f32)
  float* cb   = ws + off; off += 12 * 256;    // combined K/M biases
  // packed bf16 hi/lo weight panels (dwords). Panel layout: hi | lo.
  // slots: 0 adaptA, 1 adaptP, 2..5 qW(l,nt), 6..17 cw(l,e,kind), 18..21 aW(l,nt)
  uint32_t* pk  = (uint32_t*)(ws + off); off += 22 * 65536 / 1;  // 22 * 65536 dwords
  uint32_t* pkO = (uint32_t*)(ws + off); off += 16384;           // outW (N=64)
  float* trA = qA;  // reuse q buffers after edge phase
  float* trP = qP;

  const int st_of[3] = {1, 0, 1};  // src ntype per etype: cites, writes, written-by

  // ---- build combined relation weights: Wc = projW @ blockdiag(wrel) ----
  for (int l = 0; l < 2; ++l) {
    for (int e = 0; e < 3; ++e) {
      const int st = st_of[e];
      const int ci = l * 3 + e;
      combine_kernel<<<257, 256, 0, stream>>>(
          kW + (size_t)(l * 2 + st) * 65536, kb + (size_t)(l * 2 + st) * 256,
          watt + (size_t)ci * 16384,
          cw + (size_t)(ci * 2 + 0) * 65536, cb + (size_t)(ci * 2 + 0) * 256);
      combine_kernel<<<257, 256, 0, stream>>>(
          mW + (size_t)(l * 2 + st) * 65536, mb + (size_t)(l * 2 + st) * 256,
          wmsg + (size_t)ci * 16384,
          cw + (size_t)(ci * 2 + 1) * 65536, cb + (size_t)(ci * 2 + 1) * 256);
    }
  }

  // ---- pre-pack every weight panel into bf16 hi/lo fragment order ----
  const float* panels[22];
  panels[0] = adW;
  panels[1] = adW + 65536;
  for (int i = 0; i < 4; ++i) panels[2 + i]  = qW + (size_t)i * 65536;
  for (int i = 0; i < 12; ++i) panels[6 + i] = cw + (size_t)i * 65536;
  for (int i = 0; i < 4; ++i) panels[18 + i] = aW + (size_t)i * 65536;
  for (int p = 0; p < 22; ++p) {
    uint32_t* base = pk + (size_t)p * 65536;
    pack_weight_kernel<<<128, 256, 0, stream>>>(panels[p], base, base + 32768, 16);
  }
  pack_weight_kernel<<<32, 256, 0, stream>>>(outW, pkO, pkO + 8192, 4);

  #define PKH(slot) (pk + (size_t)(slot) * 65536)
  #define PKL(slot) (pk + (size_t)(slot) * 65536 + 32768)

  const int gA = (NA_N + 31) / 32;  // 625
  const int gP = (NP_N + 31) / 32;  // 1250

  // ---- adapt: feats = gelu(x @ adapt_W + adapt_b) ----
  wmma_gemm<256, 1><<<gA, 256, 0, stream>>>(x_a, PKH(0), PKL(0), adB, f0A, NA_N, 1.0f);
  wmma_gemm<256, 1><<<gP, 256, 0, stream>>>(x_p, PKH(1), PKL(1), adB + 256, f0P, NP_N, 1.0f);

  float* cA = f0A; float* cP = f0P;
  float* nA = f1A; float* nP = f1P;

  for (int l = 0; l < 2; ++l) {
    // Q projections per ntype
    wmma_gemm<256, 0><<<gA, 256, 0, stream>>>(
        cA, PKH(2 + l * 2 + 0), PKL(2 + l * 2 + 0),
        qb + (size_t)(l * 2 + 0) * 256, qA, NA_N, 1.0f);
    wmma_gemm<256, 0><<<gP, 256, 0, stream>>>(
        cP, PKH(2 + l * 2 + 1), PKL(2 + l * 2 + 1),
        qb + (size_t)(l * 2 + 1) * 256, qP, NP_N, 1.0f);

    hipMemsetAsync(aggA, 0, NAF * sizeof(float), stream);
    hipMemsetAsync(aggP, 0, NPF * sizeof(float), stream);

    struct Rel { int e; const float* fs; int ns; const float* q; const int* s;
                 const int* d; const float* w; float* agg; int E; };
    const Rel rels[3] = {
        {2, cP, NP_N, qA, src_wb, dst_wb, ew_wb, aggA, EWB_N},  // written-by
        {1, cA, NA_N, qP, src_w,  dst_w,  ew_w,  aggP, EW_N},   // writes
        {0, cP, NP_N, qP, src_c,  dst_c,  ew_c,  aggP, EC_N},   // cites
    };
    for (int ri = 0; ri < 3; ++ri) {
      const Rel& R = rels[ri];
      const int ci = l * 3 + R.e;
      const int gS = (R.ns + 31) / 32;
      wmma_gemm<256, 0><<<gS, 256, 0, stream>>>(
          R.fs, PKH(6 + ci * 2 + 0), PKL(6 + ci * 2 + 0),
          cb + (size_t)(ci * 2 + 0) * 256, kbuf, R.ns, 1.0f);
      wmma_gemm<256, 0><<<gS, 256, 0, stream>>>(
          R.fs, PKH(6 + ci * 2 + 1), PKL(6 + ci * 2 + 1),
          cb + (size_t)(ci * 2 + 1) * 256, mbuf, R.ns, 1.0f);
      edge_kernel<<<(R.E + 7) / 8, 256, 0, stream>>>(
          R.q, kbuf, mbuf, R.s, R.d, R.w, wpri + (size_t)(l * 3 + R.e) * 4, R.agg, R.E);
    }

    // target projection (paper agg gets the 0.5 cross-reducer mean via a_scale)
    wmma_gemm<256, 0><<<gA, 256, 0, stream>>>(
        aggA, PKH(18 + l * 2 + 0), PKL(18 + l * 2 + 0),
        ab + (size_t)(l * 2 + 0) * 256, trA, NA_N, 1.0f);
    wmma_gemm<256, 0><<<gP, 256, 0, stream>>>(
        aggP, PKH(18 + l * 2 + 1), PKL(18 + l * 2 + 1),
        ab + (size_t)(l * 2 + 1) * 256, trP, NP_N, 0.5f);

    // skip + LN -> new feats
    skip_ln_kernel<<<(NA_N + 7) / 8, 256, 0, stream>>>(
        trA, cA, skp + (size_t)l * 3 + 0, lng + (size_t)(l * 2 + 0) * 256,
        lnb + (size_t)(l * 2 + 0) * 256, nA, NA_N);
    skip_ln_kernel<<<(NP_N + 7) / 8, 256, 0, stream>>>(
        trP, cP, skp + (size_t)l * 3 + 1, lng + (size_t)(l * 2 + 1) * 256,
        lnb + (size_t)(l * 2 + 1) * 256, nP, NP_N);

    // ping-pong
    float* t;
    t = cA; cA = nA; nA = t;
    t = cP; cP = nP; nP = t;
  }

  // ---- output head: paper feats @ out_W + out_b -> [40000, 64] ----
  wmma_gemm<64, 0><<<gP, 256, 0, stream>>>(cP, pkO, pkO + 8192, outB,
                                           (float*)d_out, NP_N, 1.0f);
}
